// Interchange_76776835383950
// MI455X (gfx1250) — matrix operand order repaired and checked by reading, not compiled, not run on hardware
//
#include <hip/hip_runtime.h>

typedef __attribute__((ext_vector_type(16))) __bf16 v16bf;
typedef __attribute__((ext_vector_type(8)))  float  v8f;
typedef __attribute__((ext_vector_type(4)))  unsigned int u32x4;
typedef __attribute__((ext_vector_type(8)))  int i32x8;
typedef __attribute__((ext_vector_type(4)))  int i32x4;

#define DMODEL 64
#define NHEAD  4
#define NB     4
#define KERW   10
#define PADW   5
#define SLACK  48   // K/V buffers hold L+48 zeroed rows; data rows at [5, L+5)

union FragB {
    unsigned int u[8];
    uint4        q[2];
    v16bf        v;
};
union Pack8 {
    __bf16 e[8];
    uint4  q;
};

static __device__ __forceinline__ v8f wmma_bf16(v16bf a, v16bf b, v8f c) {
    return __builtin_amdgcn_wmma_f32_16x16x32_bf16(
        false, a, false, b, (short)0, c, false, false);
}

// A-fragment (16x32 bf16): lane = row (lane&15); half h=lane>>4 holds K runs
// [kbase+8h,+8) and [kbase+16+8h,+8)  -> two b128 loads
static __device__ __forceinline__ v16bf load_a_bf16(const __bf16* row, int kbase, int h) {
    FragB f;
    f.q[0] = *(const uint4*)(row + kbase + 8 * h);
    f.q[1] = *(const uint4*)(row + kbase + 16 + 8 * h);
    return f.v;
}

// A layout, fp32 source (convert to bf16)
static __device__ __forceinline__ v16bf load_a_f32(const float* row, int kbase, int h) {
    FragB f;
    const float* p0 = row + kbase + 8 * h;
    const float* p1 = row + kbase + 16 + 8 * h;
    float4 a0 = *(const float4*)(p0);
    float4 a1 = *(const float4*)(p0 + 4);
    float4 b0 = *(const float4*)(p1);
    float4 b1 = *(const float4*)(p1 + 4);
    f.v[0]=(__bf16)a0.x; f.v[1]=(__bf16)a0.y; f.v[2]=(__bf16)a0.z; f.v[3]=(__bf16)a0.w;
    f.v[4]=(__bf16)a1.x; f.v[5]=(__bf16)a1.y; f.v[6]=(__bf16)a1.z; f.v[7]=(__bf16)a1.w;
    f.v[8]=(__bf16)b0.x; f.v[9]=(__bf16)b0.y; f.v[10]=(__bf16)b0.z; f.v[11]=(__bf16)b0.w;
    f.v[12]=(__bf16)b1.x; f.v[13]=(__bf16)b1.y; f.v[14]=(__bf16)b1.z; f.v[15]=(__bf16)b1.w;
    return f.v;
}

// B-fragment (32x16 bf16) where the K dimension is CONTIGUOUS per lane-row:
// lane = col n; half h holds K runs [kbase+8h,+8) and [kbase+16+8h,+8).
static __device__ __forceinline__ v16bf load_b_krow(const __bf16* rowptr, int kbase, int h) {
    FragB f;
    f.q[0] = *(const uint4*)(rowptr + kbase + 8 * h);
    f.q[1] = *(const uint4*)(rowptr + kbase + 16 + 8 * h);
    return f.v;
}

// ---------------------------------------------------------------------------
// Weight transpose+convert: src fp32 (K,N) -> dst bf16 (N,K)
// ---------------------------------------------------------------------------
__global__ void wconv_kernel(const float* __restrict__ src, __bf16* __restrict__ dst,
                             int K, int N) {
    int id = blockIdx.x * blockDim.x + threadIdx.x;
    if (id >= K * N) return;
    int n = id / K, k = id - n * K;
    dst[(size_t)n * K + k] = (__bf16)src[(size_t)k * N + n];
}

// ---------------------------------------------------------------------------
// V transpose: Vp (BH, KR, 64) -> Vt (BH, 64, KR), both bf16 (pads copied too)
// ---------------------------------------------------------------------------
__global__ void vt_kernel(const __bf16* __restrict__ Vp, __bf16* __restrict__ Vt, int KR) {
    int kchunks = KR >> 3;
    int total = NB * NHEAD * DMODEL * kchunks;
    int id = blockIdx.x * blockDim.x + threadIdx.x;
    if (id >= total) return;
    int c8 = id % kchunks;
    int d  = (id / kchunks) & (DMODEL - 1);
    int bh = id / (kchunks * DMODEL);
    Pack8 p;
#pragma unroll
    for (int u = 0; u < 8; ++u)
        p.e[u] = Vp[((size_t)bh * KR + c8 * 8 + u) * DMODEL + d];
    *(uint4*)(Vt + ((size_t)bh * DMODEL + d) * KR + c8 * 8) = p.q;
}

// ---------------------------------------------------------------------------
// QKV projection: x(B,L,64) fp32 @ Wt(256,64) bf16 + bias -> bf16
// Q: (B,H,L,64).  K,V: (B,H,KR,64), data rows at l+5, rest zero.
// grid = (BL/64, 48) ; block = (32,4).  blockIdx.y = sel*16 + ntile
// ---------------------------------------------------------------------------
__global__ void proj_kernel(const float* __restrict__ x,
                            const __bf16* __restrict__ wqt, const float* __restrict__ bq,
                            const __bf16* __restrict__ wkt, const float* __restrict__ bk,
                            const __bf16* __restrict__ wvt, const float* __restrict__ bv,
                            __bf16* __restrict__ qout,
                            __bf16* __restrict__ kout,
                            __bf16* __restrict__ vout,
                            int L, int KR) {
    const int lane  = threadIdx.x;
    const int nsub  = lane & 15;
    const int h     = lane >> 4;
    const int mtile = blockIdx.x * blockDim.y + threadIdx.y;
    const int sel   = blockIdx.y >> 4;
    const int ntile = blockIdx.y & 15;
    const int row0  = mtile * 16;

    __shared__ __attribute__((aligned(16))) __bf16 stage_all[4][16][16];
    __bf16(*stage)[16] = stage_all[threadIdx.y];

    const __bf16* wt   = (sel == 0) ? wqt : (sel == 1) ? wkt : wvt;
    const float*  bias = (sel == 0) ? bq  : (sel == 1) ? bk  : bv;
    const int col  = ntile * 16 + nsub;
    const int head = ntile >> 2;

    v8f acc = {0.f,0.f,0.f,0.f,0.f,0.f,0.f,0.f};
    const float* xrow = x + (size_t)(row0 + nsub) * DMODEL;
#pragma unroll
    for (int c = 0; c < 2; ++c) {
        v16bf a = load_a_f32(xrow, c * 32, h);
        v16bf b = load_b_krow(wt + (size_t)col * DMODEL, c * 32, h);
        acc = wmma_bf16(a, b, acc);
    }
    const float bb = bias[head * DMODEL + ((ntile & 3) * 16 + nsub)];
#pragma unroll
    for (int v = 0; v < 8; ++v)
        stage[v + 8 * h][nsub] = (__bf16)(acc[v] + bb);

    // coalesced store: lane -> (row, 16B half-chunk)
    __bf16* dst = (sel == 0) ? qout : (sel == 1) ? kout : vout;
    int r    = lane >> 1;
    int half = lane & 1;
    int grow = row0 + r;
    int b_   = grow / L;
    int l    = grow - b_ * L;
    int d0   = (ntile & 3) * 16 + half * 8;
    size_t rowidx = (sel == 0) ? ((size_t)(b_ * NHEAD + head) * L + l)
                               : ((size_t)(b_ * NHEAD + head) * KR + (l + PADW));
    uint4 chunk = *(const uint4*)&stage[r][half * 8];
    *(uint4*)(dst + rowidx * DMODEL + d0) = chunk;
}

// ---------------------------------------------------------------------------
// Banded pair attention.  MODE: 0 same (11 keys), 1 down r=2, 2 up r=2.
// One wave = 16 queries of one (b,head).  grid = (Lq/64, 16); block = (32,4)
// Q tile is DMA'd into LDS by the Tensor Data Mover (with LDS padding).
// ---------------------------------------------------------------------------
template <int MODE>
__global__ void attn_kernel(const __bf16* __restrict__ Q,
                            const __bf16* __restrict__ Kp,
                            const __bf16* __restrict__ Vt,
                            __bf16* __restrict__ att,
                            int Lq, int KR, int PL, int pidx) {
    constexpr int KTILES = (MODE == 2) ? 3 : 2;
    constexpr int CHUNKS = (MODE == 2) ? 2 : 1;

    const int lane = threadIdx.x;
    const int nsub = lane & 15;
    const int h    = lane >> 4;
    const int w    = threadIdx.y;
    const int tq   = blockIdx.x * blockDim.y + w;
    const int q0   = tq * 16;
    const int bh   = blockIdx.y;
    const int b_   = bh / NHEAD;
    const int head = bh % NHEAD;

    __shared__ __attribute__((aligned(16))) __bf16 Plds_all[4][16][72];
    __shared__ __attribute__((aligned(16))) __bf16 Qlds_all[4][16][72];
    __bf16(*Plds)[72] = Plds_all[w];
    __bf16(*Qlds)[72] = Qlds_all[w];

    const __bf16* qplane = Q  + (size_t)bh * Lq * DMODEL;
    const __bf16* kplane = Kp + (size_t)bh * KR * DMODEL;
    const __bf16* vtpl   = Vt + (size_t)bh * DMODEL * KR;

    const int s0 = (MODE == 0) ? q0 : (MODE == 1) ? (q0 >> 1) : (q0 << 1);

    // ---- TDM: DMA the 16x64 bf16 Q tile into LDS, padded to 72-elem rows ----
    {
        unsigned int ldsa = (unsigned int)(uintptr_t)&Qlds[0][0];
        unsigned long long ga =
            (unsigned long long)(uintptr_t)(qplane + (size_t)q0 * DMODEL);
        unsigned int u0 = (unsigned int)__builtin_amdgcn_readfirstlane((int)ldsa);
        unsigned int u1 = (unsigned int)__builtin_amdgcn_readfirstlane((int)(ga & 0xffffffffu));
        unsigned int u2 = (unsigned int)__builtin_amdgcn_readfirstlane((int)(ga >> 32));
        // D# group0: count=1 | lds_addr | global_addr[56:0] | type=2
        u32x4 g0 = {1u, u0, u1, (u2 & 0x1ffffffu) | (2u << 30)};
        // D# group1: data_size=2B, pad_enable, pad_interval=32dw, pad_amount=4dw,
        // tensor_dim0=64, tensor_dim1=16, tile_dim0=64, tile_dim1=16, stride0=64
        i32x8 g1 = {(int)((1u << 16) | (1u << 20) | (4u << 22) | (3u << 25)),
                    (int)(64u << 16),   // tensor_dim0 lo16 -> [31:16]
                    (int)(16u << 16),   // tensor_dim1 lo16 -> [31:16]
                    (int)(64u << 16),   // tile_dim0 -> [31:16]
                    16,                 // tile_dim1
                    64,                 // tensor_dim0_stride lo32
                    0, 0};
        i32x4 z4 = {0, 0, 0, 0};
        i32x8 z8 = {0, 0, 0, 0, 0, 0, 0, 0};
        __builtin_amdgcn_tensor_load_to_lds(g0, g1, z4, z4, z8, 0);
        __builtin_amdgcn_s_wait_tensorcnt(0);
    }

    // ---- scores S = Q * K^T over the union key window ----
    v8f S[KTILES];
#pragma unroll
    for (int t = 0; t < KTILES; ++t) S[t] = (v8f){0.f,0.f,0.f,0.f,0.f,0.f,0.f,0.f};
#pragma unroll
    for (int c = 0; c < 2; ++c) {
        v16bf a = load_a_bf16(&Qlds[nsub][0], c * 32, h);
#pragma unroll
        for (int t = 0; t < KTILES; ++t) {
            int key = s0 + t * 16 + nsub;   // lands in zeroed slack when oob (masked)
            v16bf b = load_b_krow(kplane + (size_t)key * DMODEL, c * 32, h);
            S[t] = wmma_bf16(a, b, S[t]);
        }
    }

    // ---- band mask + row softmax (row = v + 8h, cols striped over 16 lanes) ----
    float inv[8];
#pragma unroll
    for (int v = 0; v < 8; ++v) {
        int row = v + 8 * h;
        int lo  = (MODE == 0) ? row : (MODE == 1) ? (row >> 1) : (row << 1);
        int hi  = lo + ((MODE == 2) ? (KERW + 1) : KERW);
        float m = -1e30f;
#pragma unroll
        for (int t = 0; t < KTILES; ++t) {
            int jj = t * 16 + nsub;
            float s = (jj >= lo && jj <= hi) ? S[t][v] : -1e30f;
            S[t][v] = s;
            m = fmaxf(m, s);
        }
        for (int o = 1; o < 16; o <<= 1) m = fmaxf(m, __shfl_xor(m, o, 16));
        float sum = 0.f;
#pragma unroll
        for (int t = 0; t < KTILES; ++t) {
            float e = __expf(S[t][v] - m);
            S[t][v] = e;
            sum += e;
        }
        for (int o = 1; o < 16; o <<= 1) sum += __shfl_xor(sum, o, 16);
        inv[v] = 1.0f / sum;
    }

    // ---- stage probabilities to LDS (zero-filled to 64 wide) ----
    unsigned int* pz = (unsigned int*)&Plds[0][0];
#pragma unroll
    for (int idx = lane; idx < 16 * 36; idx += 32) pz[idx] = 0u;
#pragma unroll
    for (int v = 0; v < 8; ++v) {
        int row = v + 8 * h;
#pragma unroll
        for (int t = 0; t < KTILES; ++t)
            Plds[row][t * 16 + nsub] = (__bf16)(S[t][v] * inv[v]);
    }

    // ---- O = P * V   (V^T rows are key-contiguous -> b128 B-fragments) ----
    v8f O[4];
#pragma unroll
    for (int nt = 0; nt < 4; ++nt) O[nt] = (v8f){0.f,0.f,0.f,0.f,0.f,0.f,0.f,0.f};
#pragma unroll
    for (int c = 0; c < CHUNKS; ++c) {
        v16bf pa = load_a_bf16(&Plds[nsub][0], c * 32, h);
#pragma unroll
        for (int nt = 0; nt < 4; ++nt) {
            const __bf16* vrow = vtpl + (size_t)(nt * 16 + nsub) * KR;
            v16bf vb = load_b_krow(vrow, s0 + c * 32, h);
            O[nt] = wmma_bf16(pa, vb, O[nt]);
        }
    }

    // ---- stage O into LDS (reuse Plds) then coalesced b128 stores ----
#pragma unroll
    for (int nt = 0; nt < 4; ++nt)
#pragma unroll
        for (int v = 0; v < 8; ++v)
            Plds[v + 8 * h][nt * 16 + nsub] = (__bf16)O[nt][v];

#pragma unroll
    for (int t = 0; t < 4; ++t) {
        int chunkid = lane + 32 * t;          // 128 chunks = 16 rows x 8
        int row = chunkid >> 3;
        int c8  = chunkid & 7;
        int l   = q0 + row;
        size_t base = ((((size_t)b_ * Lq + l) * NHEAD + head) * PL + pidx) * DMODEL;
        uint4 chunk = *(const uint4*)&Plds[row][c8 * 8];
        *(uint4*)(att + base + c8 * 8) = chunk;
    }
}

// ---------------------------------------------------------------------------
// Output projection: att(B*L, fan) bf16 @ Wct(64, fan) bf16 + bc -> out fp32
// grid = (BL/64, 4) ; block = (32,4)
// ---------------------------------------------------------------------------
__global__ void outproj_kernel(const __bf16* __restrict__ att,
                               const __bf16* __restrict__ wct,
                               const float* __restrict__ bc,
                               float* __restrict__ out,
                               int fan) {
    const int lane  = threadIdx.x;
    const int nsub  = lane & 15;
    const int h     = lane >> 4;
    const int mtile = blockIdx.x * blockDim.y + threadIdx.y;
    const int ntile = blockIdx.y;
    const int row0  = mtile * 16;

    __shared__ __attribute__((aligned(16))) float stage_all[4][16][20];
    float(*stage)[20] = stage_all[threadIdx.y];

    v8f acc = {0.f,0.f,0.f,0.f,0.f,0.f,0.f,0.f};
    const __bf16* arow = att + (size_t)(row0 + nsub) * fan;
    const __bf16* wrow = wct + (size_t)(ntile * 16 + nsub) * fan;
    const int ksteps = fan / 32;
#pragma unroll 4
    for (int c = 0; c < ksteps; ++c) {
        v16bf a = load_a_bf16(arow, c * 32, h);
        v16bf b = load_b_krow(wrow, c * 32, h);
        acc = wmma_bf16(a, b, acc);
    }
    const float bb = bc[ntile * 16 + nsub];
#pragma unroll
    for (int v = 0; v < 8; ++v)
        stage[v + 8 * h][nsub] = acc[v] + bb;

#pragma unroll
    for (int t = 0; t < 2; ++t) {
        int id  = lane + 32 * t;              // 64 chunks = 16 rows x 4
        int row = id >> 2;
        int c4  = id & 3;
        uint4 chunk = *(const uint4*)&stage[row][c4 * 4];
        *(uint4*)(out + (size_t)(row0 + row) * DMODEL + ntile * 16 + c4 * 4) = chunk;
    }
}

// ---------------------------------------------------------------------------
extern "C" void kernel_launch(void* const* d_in, const int* in_sizes, int n_in,
                              void* d_out, int out_size, void* d_ws, size_t ws_size,
                              hipStream_t stream) {
    (void)in_sizes; (void)n_in; (void)out_size; (void)ws_size;

    const float* x[3]  = {(const float*)d_in[0], (const float*)d_in[1], (const float*)d_in[2]};
    const float* wq    = (const float*)d_in[3];
    const float* bq    = (const float*)d_in[4];
    const float* wk    = (const float*)d_in[5];
    const float* bk    = (const float*)d_in[6];
    const float* wv    = (const float*)d_in[7];
    const float* bv    = (const float*)d_in[8];
    const float* wc[3] = {(const float*)d_in[9],  (const float*)d_in[11], (const float*)d_in[13]};
    const float* bc[3] = {(const float*)d_in[10], (const float*)d_in[12], (const float*)d_in[14]};
    float* out = (float*)d_out;

    const int Ls[3]  = {4096, 2048, 1024};
    const int PLs[3] = {2, 3, 2};
    int KRs[3], fans[3];
    for (int i = 0; i < 3; ++i) { KRs[i] = Ls[i] + SLACK; fans[i] = NHEAD * PLs[i] * DMODEL; }

    size_t off = 0;
    auto carve = [&](size_t elems) {
        __bf16* p = (__bf16*)((char*)d_ws + off);
        off += ((elems * sizeof(__bf16) + 255) / 256) * 256;
        return p;
    };
    __bf16 *Qb[3], *Kb[3], *Vb[3], *Vtb[3], *Ab[3];
    __bf16 *Wqt[3], *Wkt[3], *Wvt[3], *Wct[3];
    for (int i = 0; i < 3; ++i) {
        Qb[i]  = carve((size_t)NB * NHEAD * Ls[i] * DMODEL);
        Kb[i]  = carve((size_t)NB * NHEAD * KRs[i] * DMODEL);
        Vb[i]  = carve((size_t)NB * NHEAD * KRs[i] * DMODEL);
        Vtb[i] = carve((size_t)NB * NHEAD * DMODEL * KRs[i]);
        Ab[i]  = carve((size_t)NB * Ls[i] * NHEAD * PLs[i] * DMODEL);
        Wqt[i] = carve((size_t)256 * 64);
        Wkt[i] = carve((size_t)256 * 64);
        Wvt[i] = carve((size_t)256 * 64);
        Wct[i] = carve((size_t)64 * fans[i]);
    }

    // zero K/V/Vt (reference's zero padding + oob slack)
    for (int i = 0; i < 3; ++i) {
        size_t bytes = (size_t)NB * NHEAD * KRs[i] * DMODEL * sizeof(__bf16);
        (void)hipMemsetAsync(Kb[i], 0, bytes, stream);
        (void)hipMemsetAsync(Vb[i], 0, bytes, stream);
        (void)hipMemsetAsync(Vtb[i], 0, bytes, stream);
    }

    // weight transpose/convert to bf16 (N,K)
    for (int i = 0; i < 3; ++i) {
        wconv_kernel<<<64, 256, 0, stream>>>(wq + i * 64 * 256, Wqt[i], 64, 256);
        wconv_kernel<<<64, 256, 0, stream>>>(wk + i * 64 * 256, Wkt[i], 64, 256);
        wconv_kernel<<<64, 256, 0, stream>>>(wv + i * 64 * 256, Wvt[i], 64, 256);
        wconv_kernel<<<(fans[i] * 64 + 255) / 256, 256, 0, stream>>>(wc[i], Wct[i], fans[i], 64);
    }

    // QKV projections + V transpose
    for (int i = 0; i < 3; ++i) {
        proj_kernel<<<dim3(Ls[i] / 16, 48), dim3(32, 4), 0, stream>>>(
            x[i], Wqt[i], bq + i * NHEAD * 64, Wkt[i], bk + i * NHEAD * 64,
            Wvt[i], bv + i * NHEAD * 64, Qb[i], Kb[i], Vb[i], Ls[i], KRs[i]);
        int total = NB * NHEAD * DMODEL * (KRs[i] >> 3);
        vt_kernel<<<(total + 255) / 256, 256, 0, stream>>>(Vb[i], Vtb[i], KRs[i]);
    }

    // 7 tier pairs: (i, j, mode, pair-slot)
    const int pi[7] = {0, 0, 1, 1, 1, 2, 2};
    const int pj[7] = {0, 1, 0, 1, 2, 1, 2};
    const int pm[7] = {0, 1, 2, 0, 1, 2, 0};
    const int pp[7] = {0, 1, 0, 1, 2, 0, 1};
    for (int t = 0; t < 7; ++t) {
        int i = pi[t], j = pj[t];
        dim3 g(Ls[i] / 64, NB * NHEAD), b(32, 4);
        switch (pm[t]) {
        case 0: attn_kernel<0><<<g, b, 0, stream>>>(Qb[i], Kb[j], Vtb[j], Ab[i],
                                                    Ls[i], KRs[j], PLs[i], pp[t]); break;
        case 1: attn_kernel<1><<<g, b, 0, stream>>>(Qb[i], Kb[j], Vtb[j], Ab[i],
                                                    Ls[i], KRs[j], PLs[i], pp[t]); break;
        default: attn_kernel<2><<<g, b, 0, stream>>>(Qb[i], Kb[j], Vtb[j], Ab[i],
                                                     Ls[i], KRs[j], PLs[i], pp[t]); break;
        }
    }

    // output projections straight into d_out (tuple order)
    size_t ooff = 0;
    for (int i = 0; i < 3; ++i) {
        outproj_kernel<<<dim3(Ls[i] / 16, 4), dim3(32, 4), 0, stream>>>(
            Ab[i], Wct[i], bc[i], out + ooff, fans[i]);
        ooff += (size_t)NB * Ls[i] * DMODEL;
    }
}
